// QuillanGPT_17798344475060
// MI455X (gfx1250) — matrix operand order repaired and checked by reading, not compiled, not run on hardware
//
#include <hip/hip_runtime.h>
#include <math.h>

typedef __attribute__((ext_vector_type(16))) _Float16 v16h;
typedef __attribute__((ext_vector_type(8)))  float    v8f;

namespace {
constexpr int Vv = 100, Cc = 768, Hh = 32, HSs = 24, Ll = 2, Ee = 32, Tt = 422, Bb = 2;
constexpr int F4 = 4 * Cc;      // 3072
constexpr int TILE_M = 128, TILE_N = 128, TILE_K = 32;
}

// ---------------------------------------------------------------------------
// WMMA GEMM: C = epilogue(A[M,K] @ B[K,N])
// epilogue: (+bias[n]) -> (gelu) -> (*gate[m]) -> (+resid[m,n])
// f32 operands in HBM, converted to f16 while staging into double-buffered LDS.
// 256 threads = 8 waves (4 along M x 2 along N); each wave owns a 32x64 patch
// = 2x4 WMMA accumulators -> 8 v_wmma_f32_16x16x32_f16 per K-step.
// Requires: K % 32 == 0, N % 4 == 0 (true for all call sites: K in {768,3072},
// N in {768,3072,100}). OOB rows/cols are address-clamped; their products only
// reach output rows/cols that the guarded epilogue never stores.
// ---------------------------------------------------------------------------
__global__ __launch_bounds__(256)
void gemm_f16_wmma(const float* __restrict__ A, const float* __restrict__ B,
                   const float* __restrict__ bias, const float* __restrict__ resid,
                   const float* __restrict__ gate, int gate_stride, int act,
                   float* __restrict__ C, int M, int N, int K)
{
    __shared__ _Float16 As[2][TILE_M][TILE_K];       // 2 x 8 KB
    __shared__ _Float16 Bs[2][TILE_N][TILE_K + 4];   // 2 x 9 KB, transposed + padded

    const int tid   = threadIdx.x;
    const int lane  = tid & 31;
    const int wave  = tid >> 5;
    const int waveM = wave & 3;    // 4 waves along M (32 rows each)
    const int waveN = wave >> 2;   // 2 waves along N (64 cols each)
    const int m0 = blockIdx.y * TILE_M;
    const int n0 = blockIdx.x * TILE_N;
    const int half = lane >> 4;
    const int l16  = lane & 15;

    // per-thread staging coordinates (4 float4 groups each for A and B)
    int arow[4], ac4[4], bkr[4], bc4[4];
    const float* aptr[4];
    const float* bptr[4];
    #pragma unroll
    for (int i = 0; i < 4; ++i) {
        int f = tid + i * 256;
        arow[i] = f >> 3;  ac4[i] = (f & 7) * 4;     // A: 128 rows x 8 groups
        bkr[i]  = f >> 5;  bc4[i] = (f & 31) * 4;    // B: 32 k-rows x 32 groups
        int gr = m0 + arow[i]; if (gr >= M) gr = M - 1;           // clamp (safe: see header)
        aptr[i] = A + (size_t)gr * K + ac4[i];
        int gc = n0 + bc4[i];  if (gc + 3 >= N) gc = 0;           // clamp
        bptr[i] = B + (size_t)bkr[i] * N + gc;
    }

    v8f acc[2][4];
    #pragma unroll
    for (int i = 0; i < 2; ++i)
        #pragma unroll
        for (int j = 0; j < 4; ++j)
            #pragma unroll
            for (int r = 0; r < 8; ++r) acc[i][j][r] = 0.0f;

    float4 rA[4], rB[4];
    // --- preload tile 0 into registers, stage into buffer 0 ---
    #pragma unroll
    for (int i = 0; i < 4; ++i) rA[i] = *(const float4*)(aptr[i]);
    #pragma unroll
    for (int i = 0; i < 4; ++i) rB[i] = *(const float4*)(bptr[i]);
    #pragma unroll
    for (int i = 0; i < 4; ++i) {
        As[0][arow[i]][ac4[i] + 0] = (_Float16)rA[i].x;
        As[0][arow[i]][ac4[i] + 1] = (_Float16)rA[i].y;
        As[0][arow[i]][ac4[i] + 2] = (_Float16)rA[i].z;
        As[0][arow[i]][ac4[i] + 3] = (_Float16)rA[i].w;
    }
    #pragma unroll
    for (int i = 0; i < 4; ++i) {
        Bs[0][bc4[i] + 0][bkr[i]] = (_Float16)rB[i].x;
        Bs[0][bc4[i] + 1][bkr[i]] = (_Float16)rB[i].y;
        Bs[0][bc4[i] + 2][bkr[i]] = (_Float16)rB[i].z;
        Bs[0][bc4[i] + 3][bkr[i]] = (_Float16)rB[i].w;
    }
    __syncthreads();

    int buf = 0;
    for (int k0 = TILE_K; k0 < K; k0 += TILE_K) {
        // --- issue next tile's global loads (overlap with WMMA below) ---
        #pragma unroll
        for (int i = 0; i < 4; ++i) rA[i] = *(const float4*)(aptr[i] + k0);
        #pragma unroll
        for (int i = 0; i < 4; ++i) rB[i] = *(const float4*)(bptr[i] + (size_t)k0 * N);
        if (k0 + TILE_K < K) {   // speculative prefetch one more tile ahead
            __builtin_prefetch(aptr[0] + k0 + TILE_K, 0, 1);
            __builtin_prefetch(bptr[0] + (size_t)(k0 + TILE_K) * N, 0, 1);
        }

        // --- compute on current buffer ---
        {
            v16h af[2], bf[4];
            #pragma unroll
            for (int mt = 0; mt < 2; ++mt) {
                int row = waveM * 32 + mt * 16 + l16;
                #pragma unroll
                for (int i = 0; i < 8; ++i) {
                    int kb = ((i < 4) ? (2 * i) : (16 + 2 * (i - 4))) + half * 8;
                    af[mt][2 * i]     = As[buf][row][kb];
                    af[mt][2 * i + 1] = As[buf][row][kb + 1];
                }
            }
            #pragma unroll
            for (int nt = 0; nt < 4; ++nt) {
                int col = waveN * 64 + nt * 16 + l16;
                #pragma unroll
                for (int i = 0; i < 8; ++i) {
                    int kh = ((i < 4) ? (2 * i) : (16 + 2 * (i - 4))) + half * 8;
                    bf[nt][2 * i]     = Bs[buf][col][kh];
                    bf[nt][2 * i + 1] = Bs[buf][col][kh + 1];
                }
            }
            #pragma unroll
            for (int mt = 0; mt < 2; ++mt)
                #pragma unroll
                for (int nt = 0; nt < 4; ++nt)
                    acc[mt][nt] = __builtin_amdgcn_wmma_f32_16x16x32_f16(
                        false, af[mt], false, bf[nt], (short)0, acc[mt][nt], false, false);
        }

        // --- convert + stage next tile into the other buffer ---
        int nb = buf ^ 1;
        #pragma unroll
        for (int i = 0; i < 4; ++i) {
            As[nb][arow[i]][ac4[i] + 0] = (_Float16)rA[i].x;
            As[nb][arow[i]][ac4[i] + 1] = (_Float16)rA[i].y;
            As[nb][arow[i]][ac4[i] + 2] = (_Float16)rA[i].z;
            As[nb][arow[i]][ac4[i] + 3] = (_Float16)rA[i].w;
        }
        #pragma unroll
        for (int i = 0; i < 4; ++i) {
            Bs[nb][bc4[i] + 0][bkr[i]] = (_Float16)rB[i].x;
            Bs[nb][bc4[i] + 1][bkr[i]] = (_Float16)rB[i].y;
            Bs[nb][bc4[i] + 2][bkr[i]] = (_Float16)rB[i].z;
            Bs[nb][bc4[i] + 3][bkr[i]] = (_Float16)rB[i].w;
        }
        __syncthreads();
        buf = nb;
    }

    // --- tail compute on last buffer ---
    {
        v16h af[2], bf[4];
        #pragma unroll
        for (int mt = 0; mt < 2; ++mt) {
            int row = waveM * 32 + mt * 16 + l16;
            #pragma unroll
            for (int i = 0; i < 8; ++i) {
                int kb = ((i < 4) ? (2 * i) : (16 + 2 * (i - 4))) + half * 8;
                af[mt][2 * i]     = As[buf][row][kb];
                af[mt][2 * i + 1] = As[buf][row][kb + 1];
            }
        }
        #pragma unroll
        for (int nt = 0; nt < 4; ++nt) {
            int col = waveN * 64 + nt * 16 + l16;
            #pragma unroll
            for (int i = 0; i < 8; ++i) {
                int kh = ((i < 4) ? (2 * i) : (16 + 2 * (i - 4))) + half * 8;
                bf[nt][2 * i]     = Bs[buf][col][kh];
                bf[nt][2 * i + 1] = Bs[buf][col][kh + 1];
            }
        }
        #pragma unroll
        for (int mt = 0; mt < 2; ++mt)
            #pragma unroll
            for (int nt = 0; nt < 4; ++nt)
                acc[mt][nt] = __builtin_amdgcn_wmma_f32_16x16x32_f16(
                    false, af[mt], false, bf[nt], (short)0, acc[mt][nt], false, false);
    }

    // --- fused epilogue + guarded store (D layout: lane = n + (m>=8)*16, vgpr = m&7)
    #pragma unroll
    for (int mt = 0; mt < 2; ++mt) {
        #pragma unroll
        for (int nt = 0; nt < 4; ++nt) {
            int gn  = n0 + waveN * 64 + nt * 16 + l16;
            int gmb = m0 + waveM * 32 + mt * 16 + half * 8;
            if (gn < N) {
                #pragma unroll
                for (int r = 0; r < 8; ++r) {
                    int gm = gmb + r;
                    if (gm < M) {
                        float v = acc[mt][nt][r];
                        if (bias)  v += bias[gn];
                        if (act)   v = 0.5f * v * (1.0f + erff(v * 0.70710678118654752f));
                        if (gate)  v *= gate[(size_t)gm * gate_stride];
                        if (resid) v += resid[(size_t)gm * N + gn];
                        C[(size_t)gm * N + gn] = v;
                    }
                }
            }
        }
    }
}

// ---------------------------------------------------------------------------
// LayerNorm (biased variance, eps=1e-5), one token per block.
// ---------------------------------------------------------------------------
__global__ __launch_bounds__(256)
void layernorm_kernel(const float* __restrict__ x, const float* __restrict__ g,
                      const float* __restrict__ b, float* __restrict__ out, int Cdim)
{
    __shared__ float s_sum[8], s_sq[8];
    int row = blockIdx.x;
    const float* xr = x + (size_t)row * Cdim;
    float sum = 0.f, sq = 0.f;
    for (int c = threadIdx.x; c < Cdim; c += 256) { float v = xr[c]; sum += v; sq += v * v; }
    for (int off = 16; off > 0; off >>= 1) { sum += __shfl_xor(sum, off); sq += __shfl_xor(sq, off); }
    int lane = threadIdx.x & 31, wv = threadIdx.x >> 5;
    if (lane == 0) { s_sum[wv] = sum; s_sq[wv] = sq; }
    __syncthreads();
    if (wv == 0) {
        sum = (lane < 8) ? s_sum[lane] : 0.f;
        sq  = (lane < 8) ? s_sq[lane]  : 0.f;
        for (int off = 4; off > 0; off >>= 1) { sum += __shfl_xor(sum, off); sq += __shfl_xor(sq, off); }
        if (lane == 0) { s_sum[0] = sum; s_sq[0] = sq; }
    }
    __syncthreads();
    float mean = s_sum[0] / Cdim;
    float var  = s_sq[0] / Cdim - mean * mean;
    float rstd = rsqrtf(var + 1e-5f);
    for (int c = threadIdx.x; c < Cdim; c += 256)
        out[(size_t)row * Cdim + c] = (xr[c] - mean) * rstd * g[c] + b[c];
}

// ---------------------------------------------------------------------------
// Causal attention, one (b,h) per block. K tile in LDS (422*24*4 = 40 KB),
// online softmax per query row; V streamed through L2.
// q/k/v/att layout: [B, T, H*HS]
// ---------------------------------------------------------------------------
__global__ __launch_bounds__(256)
void attention_kernel(const float* __restrict__ q, const float* __restrict__ k,
                      const float* __restrict__ v, float* __restrict__ att)
{
    __shared__ float Ks[Tt * HSs];
    int b = blockIdx.x / Hh;
    int h = blockIdx.x % Hh;
    const size_t baseBT = (size_t)b * Tt;
    for (int i = threadIdx.x; i < Tt * HSs; i += 256) {
        int u = i / HSs, s = i % HSs;
        Ks[i] = k[(baseBT + u) * Cc + h * HSs + s];
    }
    __syncthreads();
    const float scale = rsqrtf((float)HSs);
    for (int t = threadIdx.x; t < Tt; t += 256) {
        float qv[HSs];
        const float* qp = q + (baseBT + t) * Cc + h * HSs;
        #pragma unroll
        for (int j = 0; j < HSs; ++j) qv[j] = qp[j];
        float m = -3.4e38f, d = 0.f, accv[HSs];
        #pragma unroll
        for (int j = 0; j < HSs; ++j) accv[j] = 0.f;
        for (int u = 0; u <= t; ++u) {
            float s = 0.f;
            const float* kr = &Ks[u * HSs];
            #pragma unroll
            for (int j = 0; j < HSs; ++j) s += qv[j] * kr[j];
            s *= scale;
            float nm = fmaxf(m, s);
            float ef = __expf(m - nm);
            float es = __expf(s - nm);
            d = d * ef + es;
            const float* vr = v + (baseBT + u) * Cc + h * HSs;
            #pragma unroll
            for (int j = 0; j < HSs; ++j) accv[j] = accv[j] * ef + es * vr[j];
            m = nm;
        }
        float inv = 1.f / d;
        float* op = att + (baseBT + t) * Cc + h * HSs;
        #pragma unroll
        for (int j = 0; j < HSs; ++j) op[j] = accv[j] * inv;
    }
}

// ---------------------------------------------------------------------------
// Gate scores + softmax over E=32 experts. One wave32 per token (lane==expert).
// ---------------------------------------------------------------------------
__global__ __launch_bounds__(256)
void gates_kernel(const float* __restrict__ h2, const float* __restrict__ gw,
                  const float* __restrict__ gb, float* __restrict__ gates, int M)
{
    int e   = threadIdx.x & 31;
    int tt  = threadIdx.x >> 5;
    int tok = blockIdx.x * 8 + tt;
    float s = -3.4e38f;
    if (tok < M) {
        s = gb[e];
        const float* hr = h2 + (size_t)tok * Cc;
        for (int c = 0; c < Cc; ++c) s += hr[c] * gw[c * Ee + e];
    }
    float mx = s;
    for (int off = 16; off > 0; off >>= 1) mx = fmaxf(mx, __shfl_xor(mx, off));
    float ex = (tok < M) ? __expf(s - mx) : 0.f;
    float sm = ex;
    for (int off = 16; off > 0; off >>= 1) sm += __shfl_xor(sm, off);
    if (tok < M) gates[(size_t)tok * Ee + e] = ex / sm;
}

// ---------------------------------------------------------------------------
// Reorder per-head weights [H][C][HS] -> [C][H*HS] so QKV is a plain GEMM.
// ---------------------------------------------------------------------------
__global__ __launch_bounds__(256)
void reorder_qkv_kernel(const float* __restrict__ W, float* __restrict__ Wr)
{
    int i = blockIdx.x * 256 + threadIdx.x;
    if (i < Hh * Cc * HSs) {
        int s = i % HSs; int rest = i / HSs; int c = rest % Cc; int h = rest / Cc;
        Wr[(size_t)c * (Hh * HSs) + h * HSs + s] = W[i];
    }
}

__global__ __launch_bounds__(256)
void embed_kernel(const int* __restrict__ idx, const float* __restrict__ tok,
                  const float* __restrict__ pos, float* __restrict__ x)
{
    int i = blockIdx.x * 256 + threadIdx.x;
    if (i < Bb * Tt * Cc) {
        int c = i % Cc; int bt = i / Cc; int t = bt % Tt;
        x[i] = tok[(size_t)idx[bt] * Cc + c] + pos[(size_t)t * Cc + c];
    }
}

// ---------------------------------------------------------------------------
extern "C" void kernel_launch(void* const* d_in, const int* in_sizes, int n_in,
                              void* d_out, int out_size, void* d_ws, size_t ws_size,
                              hipStream_t stream)
{
    const int*   idx  = (const int*)  d_in[0];
    const float* tok  = (const float*)d_in[1];
    const float* pos  = (const float*)d_in[2];
    const float* ln1g = (const float*)d_in[3];
    const float* ln1b = (const float*)d_in[4];
    const float* Wq   = (const float*)d_in[5];
    const float* Wk   = (const float*)d_in[6];
    const float* Wv   = (const float*)d_in[7];
    const float* Wo   = (const float*)d_in[8];
    const float* bo   = (const float*)d_in[9];
    const float* ln2g = (const float*)d_in[10];
    const float* ln2b = (const float*)d_in[11];
    const float* gw   = (const float*)d_in[12];
    const float* gb   = (const float*)d_in[13];
    const float* W1   = (const float*)d_in[14];
    const float* b1   = (const float*)d_in[15];
    const float* W2   = (const float*)d_in[16];
    const float* b2   = (const float*)d_in[17];
    const float* lnfg = (const float*)d_in[18];
    const float* lnfb = (const float*)d_in[19];
    const float* lmw  = (const float*)d_in[20];
    const float* lmb  = (const float*)d_in[21];
    float* out = (float*)d_out;

    const int M = Bb * Tt;                           // 844 tokens
    float* ws     = (float*)d_ws;
    float* x      = ws; ws += (size_t)M * Cc;        // residual stream
    float* hbuf   = ws; ws += (size_t)M * Cc;        // post-LN activations
    float* qb     = ws; ws += (size_t)M * Cc;
    float* kb     = ws; ws += (size_t)M * Cc;
    float* vb     = ws; ws += (size_t)M * Cc;
    float* attb   = ws; ws += (size_t)M * Cc;
    float* Wr     = ws; ws += (size_t)Cc * Cc;       // reordered QKV weight
    float* gatesb = ws; ws += (size_t)M * Ee;
    float* hid    = ws; ws += (size_t)M * F4;        // expert hidden (reused)

    dim3 blk(256);
    auto gemm = [&](const float* A, const float* B, const float* bias, const float* resid,
                    const float* gate, int gstride, int act, float* C, int Mm, int Nn, int Kk) {
        dim3 grid((Nn + TILE_N - 1) / TILE_N, (Mm + TILE_M - 1) / TILE_M);
        gemm_f16_wmma<<<grid, blk, 0, stream>>>(A, B, bias, resid, gate, gstride, act, C, Mm, Nn, Kk);
    };

    embed_kernel<<<dim3((M * Cc + 255) / 256), blk, 0, stream>>>(idx, tok, pos, x);

    const int rgrid = (Hh * Cc * HSs + 255) / 256;
    for (int l = 0; l < Ll; ++l) {
        layernorm_kernel<<<dim3(M), blk, 0, stream>>>(x, ln1g + l * Cc, ln1b + l * Cc, hbuf, Cc);

        reorder_qkv_kernel<<<dim3(rgrid), blk, 0, stream>>>(Wq + (size_t)l * Hh * Cc * HSs, Wr);
        gemm(hbuf, Wr, nullptr, nullptr, nullptr, 0, 0, qb, M, Cc, Cc);
        reorder_qkv_kernel<<<dim3(rgrid), blk, 0, stream>>>(Wk + (size_t)l * Hh * Cc * HSs, Wr);
        gemm(hbuf, Wr, nullptr, nullptr, nullptr, 0, 0, kb, M, Cc, Cc);
        reorder_qkv_kernel<<<dim3(rgrid), blk, 0, stream>>>(Wv + (size_t)l * Hh * Cc * HSs, Wr);
        gemm(hbuf, Wr, nullptr, nullptr, nullptr, 0, 0, vb, M, Cc, Cc);

        attention_kernel<<<dim3(Bb * Hh), blk, 0, stream>>>(qb, kb, vb, attb);

        // x = x + att @ Wo + bo
        gemm(attb, Wo + (size_t)l * Cc * Cc, bo + (size_t)l * Cc, x, nullptr, 0, 0, x, M, Cc, Cc);

        layernorm_kernel<<<dim3(M), blk, 0, stream>>>(x, ln2g + l * Cc, ln2b + l * Cc, hbuf, Cc);
        gates_kernel<<<dim3((M + 7) / 8), blk, 0, stream>>>(hbuf, gw + (size_t)l * Cc * Ee,
                                                            gb + (size_t)l * Ee, gatesb, M);
        for (int e = 0; e < Ee; ++e) {
            const float* W1e = W1 + (size_t)(l * Ee + e) * Cc * F4;
            const float* b1e = b1 + (size_t)(l * Ee + e) * F4;
            const float* W2e = W2 + (size_t)(l * Ee + e) * F4 * Cc;
            const float* b2e = b2 + (size_t)(l * Ee + e) * Cc;
            // hid = gelu(h2 @ W1e + b1e)
            gemm(hbuf, W1e, b1e, nullptr, nullptr, 0, 1, hid, M, F4, Cc);
            // x += gates[:,e] * (hid @ W2e + b2e)
            gemm(hid, W2e, b2e, x, gatesb + e, Ee, 0, x, M, Cc, F4);
        }
    }
    layernorm_kernel<<<dim3(M), blk, 0, stream>>>(x, lnfg, lnfb, hbuf, Cc);
    gemm(hbuf, lmw, lmb, nullptr, nullptr, 0, 0, out, M, Vv, Cc);

    (void)in_sizes; (void)n_in; (void)out_size; (void)ws_size;
}
